// GraphLP_70738111365647
// MI455X (gfx1250) — matrix operand order repaired and checked by reading, not compiled, not run on hardware
//
#include <hip/hip_runtime.h>
#include <math.h>

// ---- CDNA5 (gfx1250, wave32) fused link-prediction MLP ----
// raw = concat(zi[src], zj[dst]) [E,128] ; hid = relu(raw @ W1^T + b1) [E,128]
// score = sigmoid(hid @ W3^T + b3) ; out = [score ; score]  (neg == pos per ref)

typedef __attribute__((ext_vector_type(16))) _Float16 v16h;
typedef __attribute__((ext_vector_type(8)))  _Float16 v8h;
typedef __attribute__((ext_vector_type(4)))  _Float16 v4h;
typedef __attribute__((ext_vector_type(8)))  float    v8f;

#define DFEAT   64
#define INFEAT  128   // 2*DFEAT
#define HID     128
#define PITCH   136   // f16 per LDS row: 128 + 8 pad (row stride 272B, bank-friendly)
#define TM      128   // edges per block

__global__ __launch_bounds__(256)
void lp_fused_wmma(const float* __restrict__ zi, const float* __restrict__ zj,
                   const float* __restrict__ W1, const float* __restrict__ b1,
                   const float* __restrict__ W3, const float* __restrict__ b3,
                   const int* __restrict__ src, const int* __restrict__ dst,
                   float* __restrict__ out, int E)
{
    __shared__ __attribute__((aligned(16))) _Float16 sW[HID * PITCH];  // W1 as f16, row-major
    __shared__ __attribute__((aligned(16))) _Float16 sF[TM * PITCH];   // edge features as f16
    __shared__ float sB1[HID];
    __shared__ float sW3[HID];

    const int t = threadIdx.x;
    if (t < HID) { sB1[t] = b1[t]; sW3[t] = W3[t]; }

    // ---- Stage W1 fp32 -> f16 into LDS (thread t: row t/2, 64-float half t&1) ----
    {
        const int row = t >> 1, half = t & 1;
        const float4* s = (const float4*)(W1 + row * INFEAT + half * 64);
        _Float16* d = &sW[row * PITCH + half * 64];
        #pragma unroll
        for (int i = 0; i < 16; ++i) {
            float4 v = s[i];
            *(v4h*)(d + i * 4) =
                (v4h){(_Float16)v.x, (_Float16)v.y, (_Float16)v.z, (_Float16)v.w};
        }
    }

    // ---- Gather + concat edge features, fp32 -> f16 into LDS ----
    const int gBase = blockIdx.x * TM;
    {
        const int e = t >> 1, half = t & 1;
        int g  = gBase + e;
        int gc = (g < E) ? g : 0;                 // clamp tail: keep EXEC full
        int idx = half ? dst[gc] : src[gc];
        const float4* s = (const float4*)((half ? zj : zi) + (size_t)idx * DFEAT);
        _Float16* d = &sF[e * PITCH + half * DFEAT];
        #pragma unroll
        for (int i = 0; i < 16; ++i) {
            float4 v = s[i];
            *(v4h*)(d + i * 4) =
                (v4h){(_Float16)v.x, (_Float16)v.y, (_Float16)v.z, (_Float16)v.w};
        }
    }
    __syncthreads();

    // ---- WMMA main loop: wave owns 16 edges (M) x 128 hidden (N), K=128 ----
    const int lane = t & 31;
    const int wave = t >> 5;        // 0..7
    const int m    = lane & 15;     // A: row M ; B: col N ; C/D: col N
    const int h    = lane >> 4;     // lane half selects K sub-range

    const _Float16* aRow = &sF[(wave * 16 + m) * PITCH];

    v8f acc[8] = {};                // 8 N-tiles of 16x16 f32 accumulators

    #pragma unroll
    for (int c = 0; c < 4; ++c) {   // K chunks of 32
        // A fragment (16x32 f16): VGPR0-3 -> K = 32c+8h+[0..7], VGPR4-7 -> +16
        const int ka = c * 32 + h * 8;
        v8h alo = *(const v8h*)(aRow + ka);
        v8h ahi = *(const v8h*)(aRow + ka + 16);
        v16h a = __builtin_shufflevector(alo, ahi,
                 0,1,2,3,4,5,6,7,8,9,10,11,12,13,14,15);
        #pragma unroll
        for (int n = 0; n < 8; ++n) {
            // B fragment (32x16 f16): lane half h holds K = 32c+8h+[0..7] and +16 of col n
            const _Float16* bRow = &sW[(n * 16 + m) * PITCH];  // B[k][n] = W1[n][k]
            const int kb = c * 32 + h * 8;
            v8h blo = *(const v8h*)(bRow + kb);
            v8h bhi = *(const v8h*)(bRow + kb + 16);
            v16h bb = __builtin_shufflevector(blo, bhi,
                      0,1,2,3,4,5,6,7,8,9,10,11,12,13,14,15);
            acc[n] = __builtin_amdgcn_wmma_f32_16x16x32_f16(
                         false, a, false, bb, (short)0, acc[n], false, false);
        }
    }

    // ---- Fused epilogue: bias + relu + (hid . W3) + sigmoid ----
    // C/D layout: acc[n][r] at lane l holds element (M = r + 8*(l>>4), N = l&15)
    float partial[8];
    #pragma unroll
    for (int r = 0; r < 8; ++r) partial[r] = 0.0f;

    #pragma unroll
    for (int nt = 0; nt < 8; ++nt) {
        const float bias = sB1[nt * 16 + m];
        const float w3v  = sW3[nt * 16 + m];
        #pragma unroll
        for (int r = 0; r < 8; ++r) {
            float hv = acc[nt][r] + bias;
            hv = fmaxf(hv, 0.0f);
            partial[r] = fmaf(hv, w3v, partial[r]);
        }
    }
    // reduce the N dimension across each 16-lane half (wave32)
    #pragma unroll
    for (int r = 0; r < 8; ++r) {
        #pragma unroll
        for (int mask = 1; mask <= 8; mask <<= 1)
            partial[r] += __shfl_xor(partial[r], mask, 32);
    }

    if (m == 0) {                   // lanes 0 and 16 hold edges r and r+8
        const float bb3 = b3[0];
        const int eb = gBase + wave * 16 + h * 8;
        #pragma unroll
        for (int r = 0; r < 8; ++r) {
            int g = eb + r;
            if (g < E) {
                float sc = 1.0f / (1.0f + __expf(-(partial[r] + bb3)));
                out[g]     = sc;    // pos_score
                out[E + g] = sc;    // neg_score (same values per reference bug)
            }
        }
    }
}

extern "C" void kernel_launch(void* const* d_in, const int* in_sizes, int n_in,
                              void* d_out, int out_size, void* d_ws, size_t ws_size,
                              hipStream_t stream) {
    const float* zi = (const float*)d_in[0];
    const float* zj = (const float*)d_in[1];
    // d_in[2] = zn : unused by the reference computation
    const float* W1 = (const float*)d_in[3];
    const float* b1 = (const float*)d_in[4];
    const float* W3 = (const float*)d_in[5];
    const float* b3 = (const float*)d_in[6];
    const int* psrc = (const int*)d_in[7];
    const int* pdst = (const int*)d_in[8];
    // d_in[9], d_in[10] = neg_src/neg_dst : dead code in the reference

    const int E = in_sizes[7];
    const int blocks = (E + TM - 1) / TM;
    lp_fused_wmma<<<blocks, 256, 0, stream>>>(zi, zj, W1, b1, W3, b3,
                                              psrc, pdst, (float*)d_out, E);
}
